// ExtractModule_73065983640080
// MI455X (gfx1250) — matrix operand order repaired and checked by reading, not compiled, not run on hardware
//
#include <hip/hip_runtime.h>

// ---------------------------------------------------------------------------
// MI455X (gfx1250) implementation.
//   out[b,t] = (1/HW) * sum_{h,w} mask[t,h,w] * ifft2(|fft2(pad(x))|)[b,h,w]
// Pipeline:
//   1) row FFT (512pt Stockham, LDS)  : batch -> Freq (complex)
//   2) col FFT + magnitude (in place) : Freq  -> Freq (real in .x)
//   3) row inverse FFT (in place)     : Freq  -> Freq (complex)
//   4) col inverse FFT, real*1/HW     : Freq  -> Pseudo (f16)
//   5) fused mask-on-the-fly WMMA GEMM (split-K), B-operand staged in LDS by
//      the Tensor Data Mover (tensor_load_to_lds, double buffered,
//      s_wait_tensorcnt) -> partials
//   6) deterministic split-K reduction -> d_out [16,180]
// Workspace layout (needs ~45 MB):
//   Freq   float2[16][512][512]                @ 0          (33554432 B)
//   Pseudo _Float16[16][512][512]              @ 33554432   ( 8388608 B)
//   Part   float[SPLITK][12][16][16]           @ 41943040   ( 3145728 B)
// ---------------------------------------------------------------------------

#define PI_F 3.14159265358979323846f

typedef __attribute__((ext_vector_type(16))) _Float16     v16h;
typedef __attribute__((ext_vector_type(8)))  float        v8f;
typedef __attribute__((ext_vector_type(4)))  unsigned int v4u;
typedef __attribute__((ext_vector_type(8)))  int          v8i;
typedef __attribute__((ext_vector_type(4)))  int          v4i;

#define NB      16
#define H0      256
#define W0      256
#define HPAD    512
#define WPAD    512
#define TT      180
#define KTOT    (HPAD * WPAD)          // 262144
#define SPLITK  256
#define KSLICE  (KTOT / SPLITK)        // 1024
#define CHUNK   256                    // K staged per TDM tile (8KB LDS)
#define NCHUNK  (KSLICE / CHUNK)       // 4
#define MTILES  12                     // ceil(180/16)

#define WS_FREQ_OFF   0
#define WS_PSEUDO_OFF 33554432u
#define WS_PART_OFF   41943040u

#if defined(__has_builtin)
#  if __has_builtin(__builtin_amdgcn_tensor_load_to_lds) && \
      __has_builtin(__builtin_amdgcn_s_wait_tensorcnt)
#    define HAVE_TDM 1
#  endif
#endif
#ifndef HAVE_TDM
#  define HAVE_TDM 0
#endif

// ---------------- 512-point radix-2 Stockham FFT in LDS --------------------
// 256 threads, 1 butterfly each per stage, 9 stages, ping-pong buffers.
// sign = -1 forward, +1 inverse (unnormalized). Result lands in `b`.
__device__ __forceinline__ void fft512(float2* a, float2* b, int tid, float sign) {
  float2* src = a;
  float2* dst = b;
#pragma unroll
  for (int s = 0; s < 9; ++s) {
    int Ls = 1 << s;
    int j0 = tid & (Ls - 1);
    int j1 = tid >> s;
    float ang = sign * PI_F * (float)j0 / (float)Ls;
    float sn, cn;
    __sincosf(ang, &sn, &cn);
    float2 x = src[j1 * Ls + j0];
    float2 y = src[j1 * Ls + j0 + 256];
    float2 wy = make_float2(cn * y.x - sn * y.y, cn * y.y + sn * y.x);
    dst[(j1 * 2) * Ls + j0]      = make_float2(x.x + wy.x, x.y + wy.y);
    dst[(j1 * 2) * Ls + j0 + Ls] = make_float2(x.x - wy.x, x.y - wy.y);
    __syncthreads();
    float2* t = src; src = dst; dst = t;
  }
}

// 1) Forward FFT along w for each input row h<256 (width padded 256->512).
__global__ void row_fft_fwd(const float* __restrict__ x, float2* __restrict__ Freq) {
  __shared__ float2 bufA[512];
  __shared__ float2 bufB[512];
  int tid = threadIdx.x;                 // 0..255
  int h   = blockIdx.x;                  // 0..255
  int bb  = blockIdx.y;                  // 0..15
  const float* row = x + ((size_t)bb * H0 + h) * W0;
  bufA[tid]       = make_float2(row[tid], 0.0f);
  bufA[tid + 256] = make_float2(0.0f, 0.0f);
  __syncthreads();
  fft512(bufA, bufB, tid, -1.0f);
  float2* out = Freq + ((size_t)bb * HPAD + h) * WPAD;
  out[tid]       = bufB[tid];
  out[tid + 256] = bufB[tid + 256];
}

// 2) Forward FFT along h (rows >=256 of padded input are implicit zeros),
//    then |.|, written back in place as (mag, 0).
__global__ void col_fft_abs(float2* __restrict__ Freq) {
  __shared__ float2 bufA[512];
  __shared__ float2 bufB[512];
  int tid = threadIdx.x;
  int v   = blockIdx.x;                  // 0..511
  int bb  = blockIdx.y;
  float2* col = Freq + (size_t)bb * HPAD * WPAD + v;
  bufA[tid]       = col[(size_t)tid * WPAD];      // rows 0..255 (only ones written)
  bufA[tid + 256] = make_float2(0.0f, 0.0f);      // padded rows 256..511
  __syncthreads();
  fft512(bufA, bufB, tid, -1.0f);
  float2 z0 = bufB[tid];
  float2 z1 = bufB[tid + 256];
  col[(size_t)tid * WPAD]         = make_float2(sqrtf(z0.x * z0.x + z0.y * z0.y), 0.0f);
  col[(size_t)(tid + 256) * WPAD] = make_float2(sqrtf(z1.x * z1.x + z1.y * z1.y), 0.0f);
}

// 3) Inverse FFT along v (in place, unnormalized).
__global__ void row_ifft(float2* __restrict__ Freq) {
  __shared__ float2 bufA[512];
  __shared__ float2 bufB[512];
  int tid = threadIdx.x;
  int u   = blockIdx.x;
  int bb  = blockIdx.y;
  float2* row = Freq + ((size_t)bb * HPAD + u) * WPAD;
  bufA[tid]       = row[tid];
  bufA[tid + 256] = row[tid + 256];
  __syncthreads();
  fft512(bufA, bufB, tid, +1.0f);
  row[tid]       = bufB[tid];
  row[tid + 256] = bufB[tid + 256];
}

// 4) Inverse FFT along u; real part * 1/HW stored as f16 pseudo image.
__global__ void col_ifft_real(const float2* __restrict__ Freq, _Float16* __restrict__ Ps) {
  __shared__ float2 bufA[512];
  __shared__ float2 bufB[512];
  int tid = threadIdx.x;
  int w   = blockIdx.x;
  int bb  = blockIdx.y;
  const float2* col = Freq + (size_t)bb * HPAD * WPAD + w;
  bufA[tid]       = col[(size_t)tid * WPAD];
  bufA[tid + 256] = col[(size_t)(tid + 256) * WPAD];
  __syncthreads();
  fft512(bufA, bufB, tid, +1.0f);
  const float invHW = 1.0f / (float)KTOT;
  _Float16* out = Ps + (size_t)bb * KTOT + w;
  out[(size_t)tid * WPAD]         = (_Float16)(bufB[tid].x * invHW);
  out[(size_t)(tid + 256) * WPAD] = (_Float16)(bufB[tid + 256].x * invHW);
}

// ---------------- mask evaluation (jnp.sinc semantics) ----------------------
// mask(x) = sum_p lp[p]*(sinc(x-p)+sinc(x+p)),  sinc(y)=sin(pi y)/(pi y).
// With s0 = sin(pi x)/pi and integer p:  sinc(x-p)+sinc(x+p) = (-1)^p*2*s0*x/(x^2-p^2).
// Over a common denominator -> single v_rcp_f32:
//   mask = 2*s0*[ l0*f1*f2 + x^2*(l2*f1 - l1*f2) ] / (x*f1*f2),  f1=x^2-1, f2=x^2-4.
// Poles x ~= q in {0,+-1,+-2}: limit is 2*l0 / l1 / l2 respectively.
__device__ __forceinline__ float maskval(float x, float l0, float l1, float l2) {
  float s2  = (2.0f / PI_F) * __sinf(x * PI_F);
  float x2  = x * x;
  float f1  = x2 - 1.0f;
  float f2  = x2 - 4.0f;
  float num = l0 * f1 * f2 + x2 * (l2 * f1 - l1 * f2);
  float den = x * f1 * f2;
  float val = s2 * num * __builtin_amdgcn_rcpf(den);
  float r   = __builtin_rintf(x);
  float ar  = __builtin_fabsf(r);
  float pole = (ar < 0.5f) ? (2.0f * l0) : ((ar < 1.5f) ? l1 : l2);
  bool nearpole = (__builtin_fabsf(x - r) < 2e-6f) & (ar < 2.5f);
  return nearpole ? pole : val;
}

#if HAVE_TDM
// ---------------- Tensor Data Mover: 2D tile -> LDS -------------------------
// Tensor: pseudo f16 [16 rows(b)][262144 cols(k)], row stride 262144 elements.
// Tile:   16 x CHUNK starting at `gaddr` (tile start, per ISA D# semantics).
// LDS:    rows packed at CHUNK*2 bytes apart starting at lds_off.
// D# bitfields per CDNA5 ISA 8.3/8.4 (group0 128b, group1 256b).
__device__ __forceinline__ void tdm_load_tile(const _Float16* gaddr, unsigned lds_off) {
  unsigned long long ga = (unsigned long long)(size_t)gaddr;
  v4u g0;
  g0[0] = 1u;                                            // count=1, user mode
  g0[1] = lds_off;                                       // lds_addr (bytes)
  g0[2] = (unsigned)(ga & 0xFFFFFFFFu);                  // global_addr[31:0]
  g0[3] = (unsigned)((ga >> 32) & 0x01FFFFFFu)           // global_addr[56:32]
        | (2u << 30);                                    // type=2 ("image")
  v8i g1;
  g1[0] = (1 << 16);                                     // wg_mask=0, data_size=1 (2B)
  g1[1] = (int)((KTOT & 0xFFFF) << 16);                  // tensor_dim0[15:0] @ bits63:48
  g1[2] = (int)(((KTOT >> 16) & 0xFFFF)                  // tensor_dim0[31:16]
        | (16 << 16));                                   // tensor_dim1=16
  g1[3] = (int)(CHUNK << 16);                            // tile_dim0=CHUNK @ bits127:112
  g1[4] = 16;                                            // tile_dim1=16, tile_dim2=0
  g1[5] = (int)KTOT;                                     // tensor_dim0_stride[31:0]
  g1[6] = 0;                                             // stride0[47:32]=0, stride1 lo=0
  g1[7] = 0;
  v4i z4 = {0, 0, 0, 0};
#if __clang_major__ >= 23
  v8i z8 = {0, 0, 0, 0, 0, 0, 0, 0};
  __builtin_amdgcn_tensor_load_to_lds(g0, g1, z4, z4, z8, 0);
#else
  __builtin_amdgcn_tensor_load_to_lds(g0, g1, z4, z4, 0);
#endif
}
#endif  // HAVE_TDM

// 5) Fused split-K WMMA GEMM: D[t,b] += sum_K mask[t,K]*pseudo[b,K].
//    One wave per workgroup; grid = (12 M-tiles, SPLITK K-partitions).
//    A fragment (f16 16x32, contiguous k as B): lane L -> row t=L%16;
//      half h -> K = (L>=16?16:0) + h; masks built in-register.
//    B fragment (f16 32x16): lane L -> col b=L%16; half h -> K=(L>=16?16:0)+h
//      -> 16 contiguous f16, read from the TDM-staged LDS tile (ds_load_b128).
__global__ void masked_gemm(const _Float16* __restrict__ Ps,
                            const float* __restrict__ lp,
                            float* __restrict__ Part) {
  int lane   = threadIdx.x;            // 0..31 (wave32)
  int tTile  = blockIdx.x;             // 0..11
  int split  = blockIdx.y;             // 0..SPLITK-1
  int tLoc   = lane & 15;
  int hiHalf = lane >> 4;              // 0 or 1
  int t      = tTile * 16 + tLoc;
  float valid = (t < TT) ? 1.0f : 0.0f;        // zero padded mask rows t>=180
  float theta = PI_F * (float)t / (float)TT;   // linspace(0, pi(1-1/T), T)
  float st, ct;
  __sincosf(theta, &st, &ct);
  float l0 = lp[0], l1 = lp[1], l2 = lp[2];

  int kBeg = split * KSLICE;
  v8f acc = {0.f, 0.f, 0.f, 0.f, 0.f, 0.f, 0.f, 0.f};

#if HAVE_TDM
  __shared__ alignas(32) _Float16 stage[2][16 * CHUNK];   // double buffer, 16KB
  tdm_load_tile(Ps + kBeg, (unsigned)(size_t)(&stage[0][0]));
  for (int c = 0; c < NCHUNK; ++c) {
    if (c + 1 < NCHUNK) {
      tdm_load_tile(Ps + kBeg + (c + 1) * CHUNK,
                    (unsigned)(size_t)(&stage[(c + 1) & 1][0]));
      __builtin_amdgcn_s_wait_tensorcnt((short)1);   // chunk c landed
    } else {
      __builtin_amdgcn_s_wait_tensorcnt((short)0);
    }
    const _Float16* sb = &stage[c & 1][0];
    int kc = kBeg + c * CHUNK;
    for (int kk = 0; kk < CHUNK; kk += 32) {
      v16h bfrag = *(const v16h*)(sb + tLoc * CHUNK + kk + (hiHalf << 4));
      v16h afrag;
      int kbase = kc + kk + (hiHalf << 4);
#pragma unroll
      for (int hh = 0; hh < 16; ++hh) {
        int K  = kbase + hh;
        int hi = K >> 9;                 // image row
        int wi = K & 511;                // image col
        float m = (float)hi - ((hi >= 256) ? 512.0f : 0.0f);   // fft-wrapped index
        float n = (float)wi - ((wi >= 256) ? 512.0f : 0.0f);
        float mt = m * ct + n * st;
        afrag[hh] = (_Float16)(valid * maskval(mt, l0, l1, l2));
      }
      acc = __builtin_amdgcn_wmma_f32_16x16x32_f16(false, afrag, false, bfrag,
                                                   (short)0, acc, false, false);
    }
  }
#else
  // Fallback: B fragments straight from global (two global_load_b128 per lane).
  const _Float16* bptr = Ps + (size_t)tLoc * KTOT;
  for (int k0 = kBeg; k0 < kBeg + KSLICE; k0 += 32) {
    v16h bfrag = *(const v16h*)(bptr + k0 + (hiHalf << 4));
    v16h afrag;
    int kbase = k0 + (hiHalf << 4);
#pragma unroll
    for (int hh = 0; hh < 16; ++hh) {
      int K  = kbase + hh;
      int hi = K >> 9;
      int wi = K & 511;
      float m = (float)hi - ((hi >= 256) ? 512.0f : 0.0f);
      float n = (float)wi - ((wi >= 256) ? 512.0f : 0.0f);
      float mt = m * ct + n * st;
      afrag[hh] = (_Float16)(valid * maskval(mt, l0, l1, l2));
    }
    acc = __builtin_amdgcn_wmma_f32_16x16x32_f16(false, afrag, false, bfrag,
                                                 (short)0, acc, false, false);
  }
#endif

  // C/D f32 16x16 layout: lane L -> N=L%16; VGPR r -> M = r + (L>=16 ? 8 : 0).
  float* out = Part + ((size_t)split * MTILES + tTile) * 256;
  int N = tLoc;
  int mBase = hiHalf << 3;
#pragma unroll
  for (int r = 0; r < 8; ++r)
    out[(mBase + r) * 16 + N] = acc[r];
}

// 6) Deterministic split-K reduction; applies the einsum's 1/(H*W).
__global__ void reduce_out(const float* __restrict__ Part, float* __restrict__ out) {
  int idx = blockIdx.x * blockDim.x + threadIdx.x;
  if (idx >= NB * TT) return;
  int bb = idx / TT;                 // batch
  int t  = idx % TT;                 // theta index
  int tile = t >> 4;
  int M    = t & 15;
  float sum = 0.0f;
  for (int s = 0; s < SPLITK; ++s)
    sum += Part[((size_t)s * MTILES + tile) * 256 + M * 16 + bb];
  out[idx] = sum * (1.0f / (float)KTOT);   // out[b,t] flat = b*180 + t
}

// ---------------------------------------------------------------------------
extern "C" void kernel_launch(void* const* d_in, const int* in_sizes, int n_in,
                              void* d_out, int out_size, void* d_ws, size_t ws_size,
                              hipStream_t stream) {
  (void)in_sizes; (void)n_in; (void)out_size; (void)ws_size;
  const float* batch = (const float*)d_in[0];   // [16,256,256] f32
  const float* lp    = (const float*)d_in[1];   // [3] f32

  char* ws = (char*)d_ws;
  float2*   Freq = (float2*)(ws + WS_FREQ_OFF);
  _Float16* Ps   = (_Float16*)(ws + WS_PSEUDO_OFF);
  float*    Part = (float*)(ws + WS_PART_OFF);
  float*    out  = (float*)d_out;

  row_fft_fwd  <<<dim3(H0, NB),   256, 0, stream>>>(batch, Freq);
  col_fft_abs  <<<dim3(WPAD, NB), 256, 0, stream>>>(Freq);
  row_ifft     <<<dim3(HPAD, NB), 256, 0, stream>>>(Freq);
  col_ifft_real<<<dim3(WPAD, NB), 256, 0, stream>>>(Freq, Ps);
  masked_gemm  <<<dim3(MTILES, SPLITK), 32, 0, stream>>>(Ps, lp, Part);
  reduce_out   <<<dim3((NB * TT + 255) / 256), 256, 0, stream>>>(Part, out);
}
